// SecondOrderDeformableAlignment_22591527977131
// MI455X (gfx1250) — matrix operand order repaired and checked by reading, not compiled, not run on hardware
//
#include <hip/hip_runtime.h>
#include <math.h>

// ---------------------------------------------------------------------------
// SecondOrderDeformableAlignment on MI455X (gfx1250, wave32, WMMA)
//
// - Tap-major im2col K-ordering (K = kk*Cin + c, Cin % 32 == 0): constant tap
//   per 16x16x32 WMMA step, no divisions in the hot loop, one bounds test.
// - NHWC activations: lane A-fragments and deform corner gathers are
//   contiguous b128 loads.
// - N-blocked waves: one wave computes 16 spatial x NT*16 couts, so each A
//   fragment (and each expensive deform bilinear sample) feeds NT WMMAs.
// - Weights pre-packed to bf16 [n][kk*Cin + c] -> contiguous B fragments.
// ---------------------------------------------------------------------------

typedef __attribute__((ext_vector_type(16))) __bf16 v16bf;
typedef __attribute__((ext_vector_type(8)))  float  v8f;
typedef __attribute__((ext_vector_type(4)))  float  v4f;
typedef __attribute__((ext_vector_type(2)))  float  v2f;

#define HH  192
#define WWD 192
#define HW  (192 * 192)      // 36864 spatial positions = 2304 M-tiles of 16

__device__ __forceinline__ __bf16 to_bf16(float f) { return (__bf16)f; }

// ---------------------------------------------------------------------------
// NCHW -> NHWC transpose: dst[m*C + c] = src[c*HW + m]
// ---------------------------------------------------------------------------
__global__ __launch_bounds__(256) void nchw_to_nhwc_kernel(
    const float* __restrict__ src, float* __restrict__ dst, int C) {
  const int idx = blockIdx.x * 256 + threadIdx.x;
  if (idx >= C * HW) return;
  const int m = idx / C;
  const int c = idx - m * C;
  dst[idx] = src[(size_t)c * HW + m];
}

// ---------------------------------------------------------------------------
// Pack OIHW fp32 weights -> bf16 [n][kk*Cin + c]  (tap-major K ordering)
// ---------------------------------------------------------------------------
__global__ __launch_bounds__(256) void pack_weights_kernel(
    const float* __restrict__ src, __bf16* __restrict__ dst, int Cout, int Cin) {
  const int idx = blockIdx.x * 256 + threadIdx.x;
  if (idx >= Cout * Cin * 9) return;
  const int n  = idx / (Cin * 9);
  const int r  = idx - n * (Cin * 9);
  const int c  = r / 9;
  const int kk = r - c * 9;
  dst[(size_t)n * (Cin * 9) + kk * Cin + c] = (__bf16)src[idx];
}

// ---------------------------------------------------------------------------
// 3x3 SAME conv, NHWC -> NHWC. One wave = 16 spatial x (NT*16) couts.
//   in  : [HW, CIN] fp32        wpk : [COUT][9*CIN] bf16 (tap-major)
//   out : [HW, COUT] fp32, optional leaky-relu(0.1)
// ---------------------------------------------------------------------------
template <int CIN, int COUT, int NT, bool ACT>
__global__ __launch_bounds__(256) void conv3x3_wmma_kernel(
    const float* __restrict__ in, const __bf16* __restrict__ wpk,
    const float* __restrict__ bias, float* __restrict__ out) {
  constexpr int KTOT = CIN * 9;
  const int lane  = threadIdx.x & 31;
  const int wave  = threadIdx.x >> 5;
  const int mt    = blockIdx.x * 8 + wave;       // spatial tile
  const int lhalf = lane >> 4;
  const int l15   = lane & 15;

  const int m = mt * 16 + l15;                   // A row (spatial)
  const int h = m / WWD;
  const int w = m - h * WWD;
  const int n = blockIdx.y * (NT * 16) + l15;    // first B column of this lane

  const __bf16* __restrict__ pb = wpk + (size_t)n * KTOT + (lhalf << 4);

  v8f acc[NT];
#pragma unroll
  for (int t = 0; t < NT; ++t) acc[t] = (v8f){};

  for (int kk = 0; kk < 9; ++kk) {               // rolled: 9 cheap iterations
    const int iy = h + (kk / 3) - 1;
    const int ix = w + (kk % 3) - 1;
    const bool valid = (iy >= 0) & (iy < HH) & (ix >= 0) & (ix < WWD);
    const float* __restrict__ pa =
        in + (size_t)(iy * WWD + ix) * CIN + (lhalf << 4);
#pragma unroll
    for (int c0 = 0; c0 < CIN; c0 += 32) {
      v16bf A;
      if (valid) {
        const v4f a0 = *(const v4f*)(pa + c0);
        const v4f a1 = *(const v4f*)(pa + c0 + 4);
        const v4f a2 = *(const v4f*)(pa + c0 + 8);
        const v4f a3 = *(const v4f*)(pa + c0 + 12);
#pragma unroll
        for (int e = 0; e < 4; ++e) {
          A[e]      = to_bf16(a0[e]);
          A[e + 4]  = to_bf16(a1[e]);
          A[e + 8]  = to_bf16(a2[e]);
          A[e + 12] = to_bf16(a3[e]);
        }
      } else {
#pragma unroll
        for (int e = 0; e < 16; ++e) A[e] = to_bf16(0.0f);
      }
#pragma unroll
      for (int t = 0; t < NT; ++t) {
        // B for N-tile t: compile-time immediate offset t*16*KTOT
        const v16bf B = *(const v16bf*)(pb + (size_t)t * 16 * KTOT + kk * CIN + c0);
        acc[t] = __builtin_amdgcn_wmma_f32_16x16x32_bf16(
            false, A, false, B, (short)0, acc[t], false, false);
      }
    }
  }

  // C/D layout: col = lane&15, row = r + (lane>>4)*8 ; NHWC store
#pragma unroll
  for (int t = 0; t < NT; ++t) {
    const int nt_col = n + t * 16;
    const float bn = bias[nt_col];
#pragma unroll
    for (int r = 0; r < 8; ++r) {
      const int mr = mt * 16 + r + (lhalf << 3);
      float v = acc[t][r] + bn;
      if (ACT) v = (v >= 0.0f) ? v : 0.1f * v;
      out[(size_t)mr * COUT + nt_col] = v;
    }
  }
}

// ---------------------------------------------------------------------------
// offset/mask activation over conv4 NHWC output (in place):
//   channel c in [0,288): 5*tanh(v);  c in [288,432): sigmoid(v)
// ---------------------------------------------------------------------------
__global__ __launch_bounds__(256) void offset_mask_act_kernel(float* __restrict__ o4) {
  const int idx = blockIdx.x * 256 + threadIdx.x;
  if (idx >= 432 * HW) return;
  const int c = idx - (idx / 432) * 432;
  const float v = o4[idx];
  o4[idx] = (c < 288) ? 5.0f * tanhf(v) : 1.0f / (1.0f + __expf(-v));
}

// ---------------------------------------------------------------------------
// Deformable conv: GEMM  M=36864, N=64, K=9*128 (tap-major: K = kk*128 + c).
// One wave computes 16 spatial x all 64 couts (NT=4), so each bilinear
// sample is built once and feeds 4 WMMAs. Per 8-element A sub-group the
// deform group dg = c>>3 is constant; each bilinear corner is 8 consecutive
// NHWC channels = 2x global_load_b128.
//   xT : [HW,128] fp32 (NHWC)   wpk : [64][9*128] bf16
//   o4 : [HW,432] fp32 NHWC: c=(dg*9+kk)*2+{y,x} offsets, c=288+dg*9+kk mask
//   out: [64,HW]  fp32 (NCHW, final output)
// ---------------------------------------------------------------------------
__global__ __launch_bounds__(256) void deform_wmma_kernel(
    const float* __restrict__ xT, const __bf16* __restrict__ wpk,
    const float* __restrict__ bias, const float* __restrict__ o4,
    float* __restrict__ out) {
  constexpr int CIN = 128, KTOT = CIN * 9, NT = 4;
  const int lane  = threadIdx.x & 31;
  const int wave  = threadIdx.x >> 5;
  const int mt    = blockIdx.x * 8 + wave;
  const int lhalf = lane >> 4;
  const int l15   = lane & 15;

  const int m = mt * 16 + l15;
  const int h = m / WWD;
  const int w = m - h * WWD;
  const int n = l15;                             // first of 4 columns

  const __bf16* __restrict__ pb  = wpk + (size_t)n * KTOT + (lhalf << 4);
  const float*  __restrict__ o4m = o4 + (size_t)m * 432;

  v8f acc[NT];
#pragma unroll
  for (int t = 0; t < NT; ++t) acc[t] = (v8f){};

  for (int kk = 0; kk < 9; ++kk) {               // rolled: 9 iterations
    const float fy = (float)(h + (kk / 3) - 1);
    const float fx = (float)(w + (kk % 3) - 1);
#pragma unroll
    for (int c0 = 0; c0 < CIN; c0 += 32) {
      v16bf A;
#pragma unroll
      for (int half = 0; half < 2; ++half) {
        const int cbase = c0 + (lhalf << 4) + (half << 3);  // 8 consecutive ch
        const int dg = cbase >> 3;                           // Cg = 8
        const int dk = dg * 9 + kk;
        const v2f  oxy = *(const v2f*)(o4m + 2 * dk);        // oy, ox
        const float mk = o4m[288 + dk];
        const float py = fy + oxy[0];
        const float px = fx + oxy[1];
        const float y0f = floorf(py), x0f = floorf(px);
        const int   iy0 = (int)y0f,   ix0 = (int)x0f;
        const float wy1 = py - y0f,   wx1 = px - x0f;
        const float wy0 = 1.0f - wy1, wx0 = 1.0f - wx1;
        // branchless OOB: zero the weight, clamp the index (ref semantics)
        const float vy0 = (iy0 >= 0 && iy0 < HH)          ? 1.0f : 0.0f;
        const float vy1 = (iy0 + 1 >= 0 && iy0 + 1 < HH)  ? 1.0f : 0.0f;
        const float vx0 = (ix0 >= 0 && ix0 < WWD)         ? 1.0f : 0.0f;
        const float vx1 = (ix0 + 1 >= 0 && ix0 + 1 < WWD) ? 1.0f : 0.0f;
        const float w00 = wy0 * wx0 * vy0 * vx0 * mk;
        const float w01 = wy0 * wx1 * vy0 * vx1 * mk;
        const float w10 = wy1 * wx0 * vy1 * vx0 * mk;
        const float w11 = wy1 * wx1 * vy1 * vx1 * mk;
        const int y0c = min(max(iy0, 0), HH - 1);
        const int y1c = min(max(iy0 + 1, 0), HH - 1);
        const int x0c = min(max(ix0, 0), WWD - 1);
        const int x1c = min(max(ix0 + 1, 0), WWD - 1);
        // NHWC: 8 consecutive channels per corner, 2x b128 each
        const float* __restrict__ p00 = xT + (size_t)(y0c * WWD + x0c) * CIN + cbase;
        const float* __restrict__ p01 = xT + (size_t)(y0c * WWD + x1c) * CIN + cbase;
        const float* __restrict__ p10 = xT + (size_t)(y1c * WWD + x0c) * CIN + cbase;
        const float* __restrict__ p11 = xT + (size_t)(y1c * WWD + x1c) * CIN + cbase;
        const v4f c00a = *(const v4f*)p00, c00b = *(const v4f*)(p00 + 4);
        const v4f c01a = *(const v4f*)p01, c01b = *(const v4f*)(p01 + 4);
        const v4f c10a = *(const v4f*)p10, c10b = *(const v4f*)(p10 + 4);
        const v4f c11a = *(const v4f*)p11, c11b = *(const v4f*)(p11 + 4);
#pragma unroll
        for (int j = 0; j < 4; ++j) {
          const float sa = w00 * c00a[j] + w01 * c01a[j] +
                           w10 * c10a[j] + w11 * c11a[j];
          const float sb = w00 * c00b[j] + w01 * c01b[j] +
                           w10 * c10b[j] + w11 * c11b[j];
          A[(half << 3) + j]     = to_bf16(sa);
          A[(half << 3) + 4 + j] = to_bf16(sb);
        }
      }
#pragma unroll
      for (int t = 0; t < NT; ++t) {
        const v16bf B = *(const v16bf*)(pb + (size_t)t * 16 * KTOT + kk * CIN + c0);
        acc[t] = __builtin_amdgcn_wmma_f32_16x16x32_bf16(
            false, A, false, B, (short)0, acc[t], false, false);
      }
    }
  }

  // Final output is NCHW
#pragma unroll
  for (int t = 0; t < NT; ++t) {
    const int nt_col = n + t * 16;
    const float bn = bias[nt_col];
    float* __restrict__ ocol = out + (size_t)nt_col * HW;
#pragma unroll
    for (int r = 0; r < 8; ++r) {
      const int mr = mt * 16 + r + (lhalf << 3);
      ocol[mr] = acc[t][r] + bn;
    }
  }
}

// ---------------------------------------------------------------------------
// Launch: transpose inputs to NHWC + pack weights, then layer-by-layer
// through L2-resident workspace (~140 MB < 192 MB L2).
// ---------------------------------------------------------------------------
extern "C" void kernel_launch(void* const* d_in, const int* in_sizes, int n_in,
                              void* d_out, int out_size, void* d_ws, size_t ws_size,
                              hipStream_t stream) {
  const float* x    = (const float*)d_in[0];   // [1,128,192,192]
  const float* cond = (const float*)d_in[1];   // [1,192,192,192]
  const float* wdc  = (const float*)d_in[2];   // [64,128,3,3]
  const float* bdc  = (const float*)d_in[3];   // [64]
  const float* w1   = (const float*)d_in[4];   // [64,192,3,3]
  const float* b1   = (const float*)d_in[5];
  const float* w2   = (const float*)d_in[6];   // [64,64,3,3]
  const float* b2   = (const float*)d_in[7];
  const float* w3   = (const float*)d_in[8];   // [64,64,3,3]
  const float* b3   = (const float*)d_in[9];
  const float* w4   = (const float*)d_in[10];  // [432,64,3,3]
  const float* b4   = (const float*)d_in[11];

  float* ws    = (float*)d_ws;
  float* condT = ws;                                 // [HW,192]
  float* xT    = condT + (size_t)192 * HW;           // [HW,128]
  float* h1    = xT    + (size_t)128 * HW;           // [HW,64]
  float* h2    = h1    + (size_t)64 * HW;            // [HW,64]
  float* h3    = h2    + (size_t)64 * HW;            // [HW,64]
  float* o4    = h3    + (size_t)64 * HW;            // [HW,432]

  __bf16* pw1 = (__bf16*)(o4 + (size_t)432 * HW);    // 64*1728
  __bf16* pw2 = pw1 + (size_t)64 * 1728;             // 64*576
  __bf16* pw3 = pw2 + (size_t)64 * 576;              // 64*576
  __bf16* pw4 = pw3 + (size_t)64 * 576;              // 432*576
  __bf16* pwd = pw4 + (size_t)432 * 576;             // 64*1152

  auto cdiv = [](int a, int b) { return (a + b - 1) / b; };

  nchw_to_nhwc_kernel<<<cdiv(192 * HW, 256), 256, 0, stream>>>(cond, condT, 192);
  nchw_to_nhwc_kernel<<<cdiv(128 * HW, 256), 256, 0, stream>>>(x, xT, 128);
  pack_weights_kernel<<<cdiv(64 * 192 * 9, 256), 256, 0, stream>>>(w1, pw1, 64, 192);
  pack_weights_kernel<<<cdiv(64 * 64 * 9, 256), 256, 0, stream>>>(w2, pw2, 64, 64);
  pack_weights_kernel<<<cdiv(64 * 64 * 9, 256), 256, 0, stream>>>(w3, pw3, 64, 64);
  pack_weights_kernel<<<cdiv(432 * 64 * 9, 256), 256, 0, stream>>>(w4, pw4, 432, 64);
  pack_weights_kernel<<<cdiv(64 * 128 * 9, 256), 256, 0, stream>>>(wdc, pwd, 64, 128);

  const dim3 blk(256);                          // 8 waves/WG (wave32)
  const dim3 g64(2304 / 8, 1);                  // NT=4 -> all 64 couts/wave
  const dim3 g432(2304 / 8, 432 / 48);          // NT=3 -> 48 couts/wave

  conv3x3_wmma_kernel<192, 64, 4,  true><<<g64, blk, 0, stream>>>(condT, pw1, b1, h1);
  conv3x3_wmma_kernel<64,  64, 4,  true><<<g64, blk, 0, stream>>>(h1,    pw2, b2, h2);
  conv3x3_wmma_kernel<64,  64, 4,  true><<<g64, blk, 0, stream>>>(h2,    pw3, b3, h3);
  conv3x3_wmma_kernel<64, 432, 3, false><<<g432, blk, 0, stream>>>(h3,   pw4, b4, o4);

  offset_mask_act_kernel<<<cdiv(432 * HW, 256), 256, 0, stream>>>(o4);

  deform_wmma_kernel<<<g64, blk, 0, stream>>>(xT, pwd, bdc, o4, (float*)d_out);
}
